// CPGTaintFlowModel_48172353192361
// MI455X (gfx1250) — matrix operand order repaired and checked by reading, not compiled, not run on hardware
//
#include <hip/hip_runtime.h>
#include <hip/hip_bf16.h>

// ---------------------------------------------------------------------------
// CDNA5 (gfx1250) implementation of the CPG taint-flow GAT model.
// Dense GEMMs: v_wmma_f32_16x16x32_bf16, wave32, 16x64 output strip per wave,
// bf16 operands pre-converted/pre-packed so the K-loop is pure b128 loads+WMMA.
// Edge softmax-aggregation uses float-ordered-key atomicMax + atomicAdd f32.
// ---------------------------------------------------------------------------

#define NN 50000
#define NE 400000
#define HEADS 4
#define HID 128
#define NEG_SLOPE 0.2f

typedef __attribute__((ext_vector_type(16))) __bf16 v16bf;
typedef __attribute__((ext_vector_type(8)))  __bf16 v8bf;
typedef __attribute__((ext_vector_type(8)))  float  v8f;

__device__ __forceinline__ __bf16 f2bf(float f) {
  unsigned u = __builtin_bit_cast(unsigned, f);
  unsigned r = (u + 0x7FFFu + ((u >> 16) & 1u)) >> 16;   // round-to-nearest-even
  unsigned short s = (unsigned short)r;
  return __builtin_bit_cast(__bf16, s);
}

// monotonic float <-> uint key for atomicMax on floats
__device__ __forceinline__ unsigned fkey(float f) {
  unsigned u = __builtin_bit_cast(unsigned, f);
  return (u & 0x80000000u) ? ~u : (u | 0x80000000u);
}
__device__ __forceinline__ float funkey(unsigned k) {
  unsigned u = (k & 0x80000000u) ? (k & 0x7FFFFFFFu) : ~k;
  return __builtin_bit_cast(float, u);
}

__device__ __forceinline__ float lrelu(float x) { return x > 0.f ? x : NEG_SLOPE * x; }

// ---------------------------------------------------------------------------
// Elementwise f32 -> bf16 conversion (activations), one pass.
// ---------------------------------------------------------------------------
__global__ void cvt_bf16_kernel(const float* __restrict__ in, __bf16* __restrict__ out,
                                size_t count) {
  size_t i = (size_t)blockIdx.x * blockDim.x + threadIdx.x;
  if (i < count) out[i] = f2bf(in[i]);
}

// ---------------------------------------------------------------------------
// Pack weight B[K][N] (f32, row-major) into WMMA B-fragment order (bf16):
// out[(((tk*(N/16) + tn)*32 + lane)*16 + j], with lane = (kin>=16)*16 + (n&15),
// j = kin&15, kin = k&31, tk = k>>5, tn = n>>4.  (contiguous k order, same as the A loads.)
// Each lane's 16 elements become one contiguous 32-byte run.
// ---------------------------------------------------------------------------
__global__ void pack_b_kernel(const float* __restrict__ B, __bf16* __restrict__ out,
                              int K, int N) {
  int idx = blockIdx.x * blockDim.x + threadIdx.x;
  if (idx >= K * N) return;
  int k = idx / N, n = idx % N;
  int tk = k >> 5, kin = k & 31, tn = n >> 4, nl = n & 15;
  int lane = (kin >> 4) * 16 + nl;
  int j = kin & 15;
  out[(((size_t)tk * (N >> 4) + tn) * 32 + lane) * 16 + j] = f2bf(B[idx]);
}

// ---------------------------------------------------------------------------
// GEMM: C[M,N] = A[M,K] @ B[K,N] (+bias)(+relu).
// A: bf16 row-major.  Bp: bf16 pre-packed fragment order.  C: f32 row-major.
// One wave -> 16x64 output strip (4 accumulators, A fragment reused 4x).
// M%16==0, N%64==0, K%32==0 (all shapes here satisfy this).
// ---------------------------------------------------------------------------
__global__ void gemm_wmma_bf16(const __bf16* __restrict__ A,
                               const __bf16* __restrict__ Bp,
                               const float* __restrict__ bias,
                               float* __restrict__ C,
                               int M, int N, int K, int relu) {
  const int wave = blockIdx.x * (blockDim.x >> 5) + (threadIdx.x >> 5);
  const int lane = threadIdx.x & 31;
  const int tn = N >> 4;            // 16-wide column tiles
  const int tg = tn >> 2;           // groups of 4 column tiles
  const int tm = M >> 4;
  if (wave >= tm * tg) return;      // wave-uniform exit (EXEC stays full)
  const int g = wave % tg;
  const int tile_m = wave / tg;
  const int m0 = tile_m << 4;
  const int n0 = g << 6;
  const int half = lane >> 4;       // lane group 0/1
  const int ml = lane & 15;         // A row within tile
  const int nl = lane & 15;         // C column within tile

  v8f acc0 = {}, acc1 = {}, acc2 = {}, acc3 = {};
  const __bf16* arow = A + (size_t)(m0 + ml) * K;
  const int ktiles = K >> 5;

  for (int tk = 0; tk < ktiles; ++tk) {
    const int kk = tk << 5;
    if (tk + 1 < ktiles) __builtin_prefetch(arow + kk + 32, 0, 3);  // global_prefetch
    // A fragment: elems 0..15 -> K=kk+half*16+j (matches packed B order)
    v16bf a = *(const v16bf*)(arow + kk + half * 16);
    // B fragments: contiguous 32B per lane in packed layout
    const __bf16* bbase = Bp + (((size_t)tk * tn + (g << 2)) * 32 + lane) * 16;
    v16bf b0 = *(const v16bf*)(bbase);
    v16bf b1 = *(const v16bf*)(bbase + 32 * 16);
    v16bf b2 = *(const v16bf*)(bbase + 2 * 32 * 16);
    v16bf b3 = *(const v16bf*)(bbase + 3 * 32 * 16);
    acc0 = __builtin_amdgcn_wmma_f32_16x16x32_bf16(false, a, false, b0, (short)0, acc0, false, false);
    acc1 = __builtin_amdgcn_wmma_f32_16x16x32_bf16(false, a, false, b1, (short)0, acc1, false, false);
    acc2 = __builtin_amdgcn_wmma_f32_16x16x32_bf16(false, a, false, b2, (short)0, acc2, false, false);
    acc3 = __builtin_amdgcn_wmma_f32_16x16x32_bf16(false, a, false, b3, (short)0, acc3, false, false);
  }

  // C/D layout: lanes 0-15 rows 0..7, lanes 16-31 rows 8..15; col = lane&15
  const int row = m0 + half * 8;
  v8f accs[4] = {acc0, acc1, acc2, acc3};
#pragma unroll
  for (int tI = 0; tI < 4; ++tI) {
    const int col = n0 + tI * 16 + nl;
    const float bv = bias ? bias[col] : 0.f;
#pragma unroll
    for (int r = 0; r < 8; ++r) {
      float v = accs[tI][r] + bv;
      if (relu) v = fmaxf(v, 0.f);
      C[(size_t)(row + r) * N + col] = v;
    }
  }
}

// ---------------------------------------------------------------------------
// a_s[n,h] = sum_d xp[n,h,d]*a_src[h,d] ; same for a_d.  One block per node.
// ---------------------------------------------------------------------------
__global__ void asad_kernel(const float* __restrict__ xp,
                            const float* __restrict__ a_src,
                            const float* __restrict__ a_dst,
                            float* __restrict__ as, float* __restrict__ ad) {
  __shared__ float red[HID];
  const int n = blockIdx.x, d = threadIdx.x;
  const float* row = xp + (size_t)n * (HEADS * HID);
  for (int h = 0; h < HEADS; ++h) {
    red[d] = row[h * HID + d] * a_src[h * HID + d];
    __syncthreads();
    for (int s = HID / 2; s > 0; s >>= 1) { if (d < s) red[d] += red[d + s]; __syncthreads(); }
    if (d == 0) as[(size_t)n * HEADS + h] = red[0];
    __syncthreads();
    red[d] = row[h * HID + d] * a_dst[h * HID + d];
    __syncthreads();
    for (int s = HID / 2; s > 0; s >>= 1) { if (d < s) red[d] += red[d + s]; __syncthreads(); }
    if (d == 0) ad[(size_t)n * HEADS + h] = red[0];
    __syncthreads();
  }
}

// seed segment-max with the self-loop logit
__global__ void init_m_kernel(const float* __restrict__ as, const float* __restrict__ ad,
                              unsigned* __restrict__ mk) {
  int i = blockIdx.x * blockDim.x + threadIdx.x;
  if (i >= NN * HEADS) return;
  mk[i] = fkey(lrelu(as[i] + ad[i]));
}

__global__ void edge_max_kernel(const int* __restrict__ ei, const int* __restrict__ et,
                                const float* __restrict__ as, const float* __restrict__ ad,
                                unsigned* __restrict__ mk, int t) {
  int e = blockIdx.x * blockDim.x + threadIdx.x;
  if (e >= NE) return;
  if (t >= 0 && et[e] != t) return;
  int s = ei[e], d = ei[NE + e];
  for (int h = 0; h < HEADS; ++h) {
    float v = lrelu(as[s * HEADS + h] + ad[d * HEADS + h]);
    atomicMax(&mk[d * HEADS + h], fkey(v));
  }
}

// denom starts at exp(e_self - m)
__global__ void init_denom_kernel(const float* __restrict__ as, const float* __restrict__ ad,
                                  const unsigned* __restrict__ mk, float* __restrict__ dn) {
  int i = blockIdx.x * blockDim.x + threadIdx.x;
  if (i >= NN * HEADS) return;
  dn[i] = __expf(lrelu(as[i] + ad[i]) - funkey(mk[i]));
}

__global__ void edge_exp_kernel(const int* __restrict__ ei, const int* __restrict__ et,
                                const float* __restrict__ as, const float* __restrict__ ad,
                                const unsigned* __restrict__ mk, float* __restrict__ dn,
                                float* __restrict__ ew, int t) {
  int e = blockIdx.x * blockDim.x + threadIdx.x;
  if (e >= NE) return;
  int s = ei[e], d = ei[NE + e];
  bool on = (t < 0) || (et[e] == t);
  for (int h = 0; h < HEADS; ++h) {
    float w = 0.f;
    if (on) {
      float v = lrelu(as[s * HEADS + h] + ad[d * HEADS + h]);
      w = __expf(v - funkey(mk[d * HEADS + h]));
      atomicAdd(&dn[d * HEADS + h], w);
    }
    ew[(size_t)e * HEADS + h] = w;
  }
}

__global__ void zero_kernel(float* __restrict__ p, size_t count) {
  size_t i = (size_t)blockIdx.x * blockDim.x + threadIdx.x;
  if (i < count) p[i] = 0.f;
}

// grid = (NE, HEADS), block = HID
__global__ void edge_agg_kernel(const int* __restrict__ ei, const int* __restrict__ et,
                                const float* __restrict__ ew, const float* __restrict__ dn,
                                const float* __restrict__ xp, float* __restrict__ accum, int t) {
  int e = blockIdx.x, h = blockIdx.y, d = threadIdx.x;
  if (t >= 0 && et[e] != t) return;
  float w = ew[(size_t)e * HEADS + h];
  if (w == 0.f) return;
  int s = ei[e], dd = ei[NE + e];
  float alpha = w / dn[dd * HEADS + h];
  atomicAdd(&accum[(size_t)dd * (HEADS * HID) + h * HID + d],
            alpha * xp[(size_t)s * (HEADS * HID) + h * HID + d]);
}

// out[n, ooff+d] = scale * (mean_h(accum + alpha_self*xp) + b[d]) (opt relu)
__global__ void finalize_kernel(const float* __restrict__ accum, const float* __restrict__ xp,
                                const float* __restrict__ as, const float* __restrict__ ad,
                                const unsigned* __restrict__ mk, const float* __restrict__ dn,
                                const float* __restrict__ bvec, const float* __restrict__ etw,
                                int t, float* __restrict__ out, int ostride, int ooff, int relu) {
  int n = blockIdx.x, d = threadIdx.x;
  float sum = 0.f;
  for (int h = 0; h < HEADS; ++h) {
    float es = lrelu(as[n * HEADS + h] + ad[n * HEADS + h]);
    float aself = __expf(es - funkey(mk[n * HEADS + h])) / dn[n * HEADS + h];
    sum += accum[(size_t)n * (HEADS * HID) + h * HID + d]
         + aself * xp[(size_t)n * (HEADS * HID) + h * HID + d];
  }
  float v = sum * (1.f / HEADS) + bvec[d];
  if (etw) {  // softmax over the 4 edge-type weights, pick component t
    float w0 = etw[0], w1 = etw[1], w2 = etw[2], w3 = etw[3];
    float mx = fmaxf(fmaxf(w0, w1), fmaxf(w2, w3));
    float e0 = __expf(w0 - mx), e1 = __expf(w1 - mx), e2 = __expf(w2 - mx), e3 = __expf(w3 - mx);
    float se = e0 + e1 + e2 + e3;
    float wt = (t == 0) ? e0 : (t == 1) ? e1 : (t == 2) ? e2 : e3;
    v *= wt / se;
  }
  if (relu) v = fmaxf(v, 0.f);
  out[(size_t)n * ostride + ooff + d] = v;
}

// LayerNorm(128) + ReLU, one block per node
__global__ void ln_relu_kernel(const float* __restrict__ in, const float* __restrict__ g,
                               const float* __restrict__ b, float* __restrict__ out) {
  __shared__ float red[HID];
  int n = blockIdx.x, d = threadIdx.x;
  float x = in[(size_t)n * HID + d];
  red[d] = x; __syncthreads();
  for (int s = HID / 2; s > 0; s >>= 1) { if (d < s) red[d] += red[d + s]; __syncthreads(); }
  float mu = red[0] * (1.f / HID);
  __syncthreads();
  float c = x - mu;
  red[d] = c * c; __syncthreads();
  for (int s = HID / 2; s > 0; s >>= 1) { if (d < s) red[d] += red[d + s]; __syncthreads(); }
  float var = red[0] * (1.f / HID);
  float y = c * rsqrtf(var + 1e-5f) * g[d] + b[d];
  out[(size_t)n * HID + d] = fmaxf(y, 0.f);
}

// per-node 4-head x 32-dim attention; writes x_sem into fused[:,128:256]
__global__ void node_attn_kernel(const float* __restrict__ q, const float* __restrict__ k,
                                 const float* __restrict__ v, float* __restrict__ fused) {
  __shared__ float sq[HID], sk[HID], sv[HID];
  int n = blockIdx.x, d = threadIdx.x;
  sq[d] = q[(size_t)n * HID + d];
  sk[d] = k[(size_t)n * HID + d];
  sv[d] = v[(size_t)n * HID + d];
  __syncthreads();
  int h = d >> 5, dd = d & 31;
  float sc[4];
  float mx = -1e30f;
  for (int g = 0; g < 4; ++g) {
    float s = 0.f;
    for (int j = 0; j < 32; ++j) s += sq[h * 32 + j] * sk[g * 32 + j];
    s *= 0.17677669529663687f;  // 1/sqrt(32)
    sc[g] = s; mx = fmaxf(mx, s);
  }
  float se = 0.f;
  for (int g = 0; g < 4; ++g) { sc[g] = __expf(sc[g] - mx); se += sc[g]; }
  float o = 0.f;
  for (int g = 0; g < 4; ++g) o += sc[g] * sv[g * 32 + dd];
  fused[(size_t)n * 256 + 128 + h * 32 + dd] = o / se;
}

__global__ void copy_struct_kernel(const float* __restrict__ xs, float* __restrict__ fused) {
  size_t i = (size_t)blockIdx.x * blockDim.x + threadIdx.x;
  if (i >= (size_t)NN * HID) return;
  size_t n = i / HID, d = i % HID;
  fused[n * 256 + d] = xs[i];
}

// ---------------------------------------------------------------------------
static inline void launch_gemm(const __bf16* A, const __bf16* Bp, const float* bias,
                               float* C, int M, int N, int K, int relu, hipStream_t s) {
  int waves = (M / 16) * (N / 64);
  int blocks = (waves + 7) / 8;
  gemm_wmma_bf16<<<blocks, 256, 0, s>>>(A, Bp, bias, C, M, N, K, relu);
}
static inline void launch_pack(const float* W, __bf16* out, int K, int N, hipStream_t s) {
  int cnt = K * N;
  pack_b_kernel<<<(cnt + 255) / 256, 256, 0, s>>>(W, out, K, N);
}
static inline void launch_cvt(const float* in, __bf16* out, size_t cnt, hipStream_t s) {
  cvt_bf16_kernel<<<(int)((cnt + 255) / 256), 256, 0, s>>>(in, out, cnt);
}

extern "C" void kernel_launch(void* const* d_in, const int* in_sizes, int n_in,
                              void* d_out, int out_size, void* d_ws, size_t ws_size,
                              hipStream_t stream) {
  (void)in_sizes; (void)n_in; (void)out_size; (void)ws_size;
  const float* x        = (const float*)d_in[0];
  const float* codebert = (const float*)d_in[1];
  const int*   ei       = (const int*)d_in[2];    // (2, NE)
  const int*   et       = (const int*)d_in[3];    // (NE,)
  // etype_gats[t]: W, a_src, a_dst, b  at indices 4+4t .. 7+4t
  const float* etw      = (const float*)d_in[20]; // etype_w (4,)
  const float* fus_w    = (const float*)d_in[21];
  const float* fus_b    = (const float*)d_in[22];
  const float* eln_g    = (const float*)d_in[23];
  const float* eln_b    = (const float*)d_in[24];
  const float* daf_W    = (const float*)d_in[25];
  const float* daf_asrc = (const float*)d_in[26];
  const float* daf_adst = (const float*)d_in[27];
  const float* daf_b    = (const float*)d_in[28];
  const float* q_w      = (const float*)d_in[29];
  const float* q_b      = (const float*)d_in[30];
  const float* k_w      = (const float*)d_in[31];
  const float* k_b      = (const float*)d_in[32];
  const float* v_w      = (const float*)d_in[33];
  const float* v_b      = (const float*)d_in[34];
  const float* dfu_w    = (const float*)d_in[35];
  const float* dfu_b    = (const float*)d_in[36];
  const float* dln_g    = (const float*)d_in[37];
  const float* dln_b    = (const float*)d_in[38];

  // ---------------- workspace layout (f32 units unless noted) ----------------
  float* ws = (float*)d_ws;
  float* xp    = ws;                                   // N*512
  float* accum = xp    + (size_t)NN * 512;             // N*512
  float* hcat  = accum + (size_t)NN * 512;             // N*512
  float* hbuf  = hcat  + (size_t)NN * 512;             // N*128
  float* xs    = hbuf  + (size_t)NN * HID;             // N*128
  float* fused = xs    + (size_t)NN * HID;             // N*256
  float* asb   = fused + (size_t)NN * 256;             // N*4
  float* adb   = asb   + (size_t)NN * HEADS;           // N*4
  unsigned* mk = (unsigned*)(adb + (size_t)NN * HEADS);// N*4
  float* dnb   = (float*)(mk + (size_t)NN * HEADS);    // N*4
  float* ewb   = dnb + (size_t)NN * HEADS;             // E*4
  // bf16 arenas
  __bf16* xb   = (__bf16*)(ewb + (size_t)NE * HEADS);  // N*128 bf16
  __bf16* pwa  = xb + (size_t)NN * HID;                // packed weights arena
  __bf16* pW[4];
  __bf16* p = pwa;
  for (int t = 0; t < 4; ++t) { pW[t] = p; p += (size_t)HID * 512; }
  __bf16* pFus = p; p += (size_t)512 * HID;
  __bf16* pDaf = p; p += (size_t)HID * 512;
  __bf16* pQ   = p; p += (size_t)HID * HID;
  __bf16* pK   = p; p += (size_t)768 * HID;
  __bf16* pV   = p; p += (size_t)768 * HID;
  __bf16* pDfu = p; p += (size_t)256 * HID;
  // aliases (sequencing on the single stream makes these safe):
  __bf16* abuf = (__bf16*)accum;                 // bf16 A-staging (<= N*512 bf16)
  __bf16* cbb  = (__bf16*)xp;                    // codebert bf16 (N*768 bf16)
  float*  tmp2 = xp + (size_t)NN * 384;          // pre-LN daf-fusion out (N*128)
  float*  tmp1 = fused;                          // pre-LN etype-fusion out (N*128)
  float* qb = hcat;                              // q/k/v reuse hcat region
  float* kb = hcat + (size_t)NN * HID;
  float* vb = hcat + 2 * (size_t)NN * HID;

  const int nh_blk = (NN * HEADS + 255) / 256;
  const int e_blk  = (NE + 255) / 256;
  const size_t acc_cnt = (size_t)NN * 512;
  const int acc_blk = (int)((acc_cnt + 255) / 256);
  dim3 agg_grid(NE, HEADS);

  // ---- one-time conversions / weight packing ----
  launch_cvt(x, xb, (size_t)NN * HID, stream);
  for (int t = 0; t < 4; ++t) launch_pack((const float*)d_in[4 + 4 * t], pW[t], HID, 512, stream);
  launch_pack(fus_w, pFus, 512, HID, stream);
  launch_pack(daf_W, pDaf, HID, 512, stream);
  launch_pack(q_w, pQ, HID, HID, stream);
  launch_pack(k_w, pK, 768, HID, stream);
  launch_pack(v_w, pV, 768, HID, stream);
  launch_pack(dfu_w, pDfu, 256, HID, stream);

  // ---- EdgeTypeAwareGATLayer: 4 masked GATConvs -> hcat (N, 512) ----
  for (int t = 0; t < 4; ++t) {
    const float* asrc = (const float*)d_in[5 + 4 * t];
    const float* adst = (const float*)d_in[6 + 4 * t];
    const float* bv   = (const float*)d_in[7 + 4 * t];
    launch_gemm(xb, pW[t], nullptr, xp, NN, 512, HID, 0, stream);
    asad_kernel<<<NN, HID, 0, stream>>>(xp, asrc, adst, asb, adb);
    init_m_kernel<<<nh_blk, 256, 0, stream>>>(asb, adb, mk);
    edge_max_kernel<<<e_blk, 256, 0, stream>>>(ei, et, asb, adb, mk, t);
    init_denom_kernel<<<nh_blk, 256, 0, stream>>>(asb, adb, mk, dnb);
    edge_exp_kernel<<<e_blk, 256, 0, stream>>>(ei, et, asb, adb, mk, dnb, ewb, t);
    zero_kernel<<<acc_blk, 256, 0, stream>>>(accum, acc_cnt);
    edge_agg_kernel<<<agg_grid, HID, 0, stream>>>(ei, et, ewb, dnb, xp, accum, t);
    finalize_kernel<<<NN, HID, 0, stream>>>(accum, xp, asb, adb, mk, dnb, bv, etw,
                                            t, hcat, 512, t * HID, 0);
  }
  // fusion: h = relu(LN(hcat @ fus_w + fus_b))
  launch_cvt(hcat, abuf, (size_t)NN * 512, stream);
  launch_gemm(abuf, pFus, fus_b, tmp1, NN, HID, 512, 0, stream);
  ln_relu_kernel<<<NN, HID, 0, stream>>>(tmp1, eln_g, eln_b, hbuf);

  // ---- DAF GAT over all edges: x_struct = relu(gat(hbuf)) ----
  launch_cvt(hbuf, abuf, (size_t)NN * HID, stream);
  launch_gemm(abuf, pDaf, nullptr, xp, NN, 512, HID, 0, stream);
  asad_kernel<<<NN, HID, 0, stream>>>(xp, daf_asrc, daf_adst, asb, adb);
  init_m_kernel<<<nh_blk, 256, 0, stream>>>(asb, adb, mk);
  edge_max_kernel<<<e_blk, 256, 0, stream>>>(ei, et, asb, adb, mk, -1);
  init_denom_kernel<<<nh_blk, 256, 0, stream>>>(asb, adb, mk, dnb);
  edge_exp_kernel<<<e_blk, 256, 0, stream>>>(ei, et, asb, adb, mk, dnb, ewb, -1);
  zero_kernel<<<acc_blk, 256, 0, stream>>>(accum, acc_cnt);
  edge_agg_kernel<<<agg_grid, HID, 0, stream>>>(ei, et, ewb, dnb, xp, accum, -1);
  finalize_kernel<<<NN, HID, 0, stream>>>(accum, xp, asb, adb, mk, dnb, daf_b, nullptr,
                                          -1, xs, HID, 0, 1);

  // ---- q/k/v projections ----
  launch_cvt(xs, abuf, (size_t)NN * HID, stream);
  launch_gemm(abuf, pQ, q_b, qb, NN, HID, HID, 0, stream);
  launch_cvt(codebert, cbb, (size_t)NN * 768, stream);  // xp region now free
  launch_gemm(cbb, pK, k_b, kb, NN, HID, 768, 0, stream);
  launch_gemm(cbb, pV, v_b, vb, NN, HID, 768, 0, stream);

  // ---- per-node attention + fuse ----
  node_attn_kernel<<<NN, HID, 0, stream>>>(qb, kb, vb, fused);
  copy_struct_kernel<<<(NN * HID + 255) / 256, 256, 0, stream>>>(xs, fused);

  // ---- final fusion: out = relu(LN(fused @ dfu_w + dfu_b)) ----
  launch_cvt(fused, abuf, (size_t)NN * 256, stream);
  launch_gemm(abuf, pDfu, dfu_b, tmp2, NN, HID, 256, 0, stream);
  ln_relu_kernel<<<NN, HID, 0, stream>>>(tmp2, dln_g, dln_b, (float*)d_out);
}
